// SequencedLSTMs_49100066128475
// MI455X (gfx1250) — matrix operand order repaired and checked by reading, not compiled, not run on hardware
//
#include <hip/hip_runtime.h>
#include <hip/hip_bf16.h>

// ---------------- problem dims ----------------
#define Bdim 128
#define Tdim 512
#define Hdim 1024
#define G4   4096      // 4*H gates
#define NBLK 256       // persistent grid blocks
#define NTHR 256       // 8 wave32 per block

typedef __attribute__((ext_vector_type(8)))  float  v8f;
typedef __attribute__((ext_vector_type(16))) __bf16 v16bf;
typedef __attribute__((ext_vector_type(8)))  __bf16 v8bf;

// ---------------- workspace layout (bytes) ----------------
static constexpr size_t SZ_W     = (size_t)G4 * Hdim * 2;     // one bf16 weight matrix
static constexpr size_t OFF_CNT  = 0;                         // grid-barrier counter
static constexpr size_t OFF_WHH1 = 256;
static constexpr size_t OFF_WIH2 = OFF_WHH1 + SZ_W;
static constexpr size_t OFF_WHH2 = OFF_WIH2 + SZ_W;
static constexpr size_t OFF_H1   = OFF_WHH2 + SZ_W;           // B*H bf16
static constexpr size_t OFF_H2   = OFF_H1 + (size_t)Bdim*Hdim*2;
static constexpr size_t OFF_C1   = OFF_H2 + (size_t)Bdim*Hdim*2;   // B*H f32
static constexpr size_t OFF_C2   = OFF_C1 + (size_t)Bdim*Hdim*4;
static constexpr size_t OFF_RAW1 = OFF_C2 + (size_t)Bdim*Hdim*4;   // B*G4 f32
static constexpr size_t OFF_R2IH = OFF_RAW1 + (size_t)Bdim*G4*4;
static constexpr size_t OFF_R2HH = OFF_R2IH + (size_t)Bdim*G4*4;

// ---------------- helpers ----------------
__device__ __forceinline__ unsigned short f32_to_bf16_rne(float f) {
    unsigned u = __float_as_uint(f);
    unsigned r = u + 0x7FFFu + ((u >> 16) & 1u);
    return (unsigned short)(r >> 16);
}

__device__ __forceinline__ float sigmoidf_(float x) {
    return 1.0f / (1.0f + __expf(-x));
}

// device-wide barrier for a persistent co-resident grid (monotonic counter,
// reset by init kernel each launch -> graph-replay safe)
__device__ __forceinline__ void grid_sync(unsigned* cnt, unsigned phases_done) {
    __syncthreads();
    if (threadIdx.x == 0) {
        __threadfence();
        __hip_atomic_fetch_add(cnt, 1u, __ATOMIC_RELEASE, __HIP_MEMORY_SCOPE_AGENT);
        const unsigned target = phases_done * (unsigned)gridDim.x;
        while (__hip_atomic_load(cnt, __ATOMIC_ACQUIRE, __HIP_MEMORY_SCOPE_AGENT) < target)
            __builtin_amdgcn_s_sleep(1);
    }
    __syncthreads();
}

// deterministic 256-thread tree reduction
__device__ __forceinline__ float block_sum(float v, float* red) {
    const int tid = threadIdx.x;
    red[tid] = v;
    __syncthreads();
    for (int s = NTHR / 2; s > 0; s >>= 1) {
        if (tid < s) red[tid] += red[tid + s];
        __syncthreads();
    }
    float r = red[0];
    __syncthreads();
    return r;
}

// ---------------- WMMA fragment loaders ----------------
// A 16x32 fragment: lane r, element i -> K = half*16+i (matches B).
// B 32x16 as loaded here (not the ISA table order): lane n, element i -> K = half*16+i.
__device__ __forceinline__ v16bf load_a_frag(const __bf16* Arow, int k0, int half) {
    return *(const v16bf*)(Arow + k0 + half * 16);
}

__device__ __forceinline__ void store_d_tile(const v8f& acc, float* raw,
                                             int m0, int n0, int r, int half) {
    // C/D layout: VGPR v, lanes 0-15 -> M=v,N=lane ; lanes 16-31 -> M=v+8,N=lane-16
    float* out = raw + (size_t)(m0 + half * 8) * G4 + n0 + r;
    #pragma unroll
    for (int v = 0; v < 8; ++v) out[(size_t)v * G4] = acc[v];
}

// one 16x16 output tile of raw = A(BxH) * W(G4xH)^T
__device__ __forceinline__ void wmma_tile_bf16(
    const unsigned short* __restrict__ Abf,   // Bdim x Hdim
    const unsigned short* __restrict__ Wbf,   // G4 x Hdim
    float* __restrict__ raw,                  // Bdim x G4
    int m0, int n0, int lane)
{
    const int r    = lane & 15;
    const int half = lane >> 4;
    const __bf16* Arow = (const __bf16*)(Abf + (size_t)(m0 + r) * Hdim);
    const __bf16* Wrow = (const __bf16*)(Wbf + (size_t)(n0 + r) * Hdim);

    v8f acc = {};
    #pragma unroll 4
    for (int k0 = 0; k0 < Hdim; k0 += 32) {
        v16bf a = load_a_frag(Arow, k0, half);
        v16bf b = *(const v16bf*)(Wrow + k0 + half * 16);
        acc = __builtin_amdgcn_wmma_f32_16x16x32_bf16(false, a, false, b,
                                                      (short)0, acc, false, false);
    }
    store_d_tile(acc, raw, m0, n0, r, half);
}

// two adjacent-N 16x16 tiles sharing one A fragment (half-grid hidden GEMMs)
__device__ __forceinline__ void wmma_tile_bf16_x2(
    const unsigned short* __restrict__ Abf,
    const unsigned short* __restrict__ Wbf,
    float* __restrict__ raw,
    int m0, int n0, int lane)                 // covers N columns [n0, n0+32)
{
    const int r    = lane & 15;
    const int half = lane >> 4;
    const __bf16* Arow  = (const __bf16*)(Abf + (size_t)(m0 + r) * Hdim);
    const __bf16* Wrow0 = (const __bf16*)(Wbf + (size_t)(n0 + r) * Hdim);
    const __bf16* Wrow1 = (const __bf16*)(Wbf + (size_t)(n0 + 16 + r) * Hdim);

    v8f acc0 = {}, acc1 = {};
    #pragma unroll 2
    for (int k0 = 0; k0 < Hdim; k0 += 32) {
        v16bf a  = load_a_frag(Arow, k0, half);          // loaded once, used twice
        v16bf b0 = *(const v16bf*)(Wrow0 + k0 + half * 16);
        v16bf b1 = *(const v16bf*)(Wrow1 + k0 + half * 16);
        acc0 = __builtin_amdgcn_wmma_f32_16x16x32_bf16(false, a, false, b0,
                                                       (short)0, acc0, false, false);
        acc1 = __builtin_amdgcn_wmma_f32_16x16x32_bf16(false, a, false, b1,
                                                       (short)0, acc1, false, false);
    }
    store_d_tile(acc0, raw, m0, n0,      r, half);
    store_d_tile(acc1, raw, m0, n0 + 16, r, half);
}

// ---------------- LayerNorm-LSTM cell for one batch row (one workgroup) ----------------
__device__ void lstm_elemwise(
    int b,
    const float* __restrict__ rawhh,      // B x G4 : h @ Whh^T (raw)
    const float* __restrict__ rawih,      // B x G4 : x @ Wih^T (raw), or null (layer 1)
    float xw,                             // layer 1: scalar x[b,t]; row = xw * Wih1[j]
    const float* __restrict__ Wih1,       // layer 1 only (G4 x 1)
    const float* __restrict__ gih, const float* __restrict__ bih,
    const float* __restrict__ ghh, const float* __restrict__ bhh,
    const float* __restrict__ bb,
    const float* __restrict__ gc,  const float* __restrict__ bc,
    float* __restrict__ cbuf,             // B x H  (f32 cell state)
    unsigned short* __restrict__ hbf,     // B x H  (bf16 hidden, next-step WMMA A)
    float* __restrict__ outacc,           // B x H  running sum of h2, or null
    float* s_hh, float* s_ih, float* s_red)
{
    const int tid = threadIdx.x;

    // stage raw gate rows into LDS + row stats (LN over all 4096 gates)
    float shh = 0.f, qhh = 0.f, sih = 0.f, qih = 0.f;
    for (int j = tid; j < G4; j += NTHR) {
        float vh = rawhh[(size_t)b * G4 + j];
        float vi = rawih ? rawih[(size_t)b * G4 + j] : xw * Wih1[j];
        s_hh[j] = vh; s_ih[j] = vi;
        shh += vh; qhh += vh * vh;
        sih += vi; qih += vi * vi;
    }
    shh = block_sum(shh, s_red); qhh = block_sum(qhh, s_red);
    sih = block_sum(sih, s_red); qih = block_sum(qih, s_red);
    const float inv = 1.0f / (float)G4;
    const float mhh = shh * inv, mih = sih * inv;
    const float rhh = rsqrtf(fmaxf(qhh * inv - mhh * mhh, 0.f) + 1e-5f);
    const float rih = rsqrtf(fmaxf(qih * inv - mih * mih, 0.f) + 1e-5f);

    // gates -> cell update (4 hidden units per thread)
    float cr[4], og_[4];
    float sc = 0.f, qc = 0.f;
    #pragma unroll
    for (int q = 0; q < 4; ++q) {
        const int k = tid + q * NTHR;           // 0..1023
        float pre[4];
        #pragma unroll
        for (int g = 0; g < 4; ++g) {           // i, f, g, o
            const int j = g * Hdim + k;
            float hhn = (s_hh[j] - mhh) * rhh * ghh[j] + bhh[j];
            float ihn = (s_ih[j] - mih) * rih * gih[j] + bih[j];
            pre[g] = hhn + ihn + bb[j];
        }
        const float ig = sigmoidf_(pre[0]);
        const float fg = sigmoidf_(pre[1]);
        const float gg = tanhf(pre[2]);
        og_[q] = sigmoidf_(pre[3]);
        float c = fg * cbuf[(size_t)b * Hdim + k] + ig * gg;
        cbuf[(size_t)b * Hdim + k] = c;
        cr[q] = c;
        sc += c; qc += c * c;
    }

    // LN over cell state, then h
    sc = block_sum(sc, s_red); qc = block_sum(qc, s_red);
    const float mc = sc / (float)Hdim;
    const float rc = rsqrtf(fmaxf(qc / (float)Hdim - mc * mc, 0.f) + 1e-5f);
    #pragma unroll
    for (int q = 0; q < 4; ++q) {
        const int k = tid + q * NTHR;
        float lnc = (cr[q] - mc) * rc * gc[k] + bc[k];
        float h   = og_[q] * tanhf(lnc);
        hbf[(size_t)b * Hdim + k] = f32_to_bf16_rne(h);
        if (outacc) outacc[(size_t)b * Hdim + k] += h;
    }
}

// ---------------- init: bf16 weight copies, state init, out=0, barrier=0 ----------------
__global__ void init_kernel(const float* __restrict__ Whh1, const float* __restrict__ Wih2,
                            const float* __restrict__ Whh2,
                            const float* __restrict__ h01, const float* __restrict__ c01,
                            const float* __restrict__ h02, const float* __restrict__ c02,
                            char* __restrict__ ws, float* __restrict__ out)
{
    unsigned short* wh1 = (unsigned short*)(ws + OFF_WHH1);
    unsigned short* wi2 = (unsigned short*)(ws + OFF_WIH2);
    unsigned short* wh2 = (unsigned short*)(ws + OFF_WHH2);
    unsigned short* h1  = (unsigned short*)(ws + OFF_H1);
    unsigned short* h2  = (unsigned short*)(ws + OFF_H2);
    float* c1p = (float*)(ws + OFF_C1);
    float* c2p = (float*)(ws + OFF_C2);

    const size_t NW = (size_t)G4 * Hdim;
    const size_t i  = (size_t)blockIdx.x * blockDim.x + threadIdx.x;
    const size_t st = (size_t)gridDim.x * blockDim.x;

    for (size_t idx = i; idx < 3 * NW; idx += st) {
        if (idx < NW)            wh1[idx]          = f32_to_bf16_rne(Whh1[idx]);
        else if (idx < 2 * NW)   wi2[idx - NW]     = f32_to_bf16_rne(Wih2[idx - NW]);
        else                     wh2[idx - 2 * NW] = f32_to_bf16_rne(Whh2[idx - 2 * NW]);
    }
    const size_t NH = (size_t)Bdim * Hdim;
    for (size_t idx = i; idx < NH; idx += st) {
        h1[idx]  = f32_to_bf16_rne(h01[idx]);
        h2[idx]  = f32_to_bf16_rne(h02[idx]);
        c1p[idx] = c01[idx];
        c2p[idx] = c02[idx];
        out[idx] = 0.0f;
    }
    if (i == 0) *(unsigned*)(ws + OFF_CNT) = 0u;
}

// ---------------- persistent 2-layer LN-LSTM over T steps ----------------
// Steady state per step (3 grid barriers, 2 of 3 GEMMs hidden under cell math):
//   P2: WG<128 elemwise1 -> h1_t     || WG>=128 GEMM r2hh = h2_{t-1} @ Whh2^T
//   P3: all WGs GEMM r2ih = h1_t @ Wih2^T
//   P4: WG<128 elemwise2 -> h2_t,out || WG>=128 GEMM raw1 = h1_t @ Whh1^T (for t+1)
__global__ void __launch_bounds__(NTHR, 1)
lstm_persistent(const float* __restrict__ x,
                const float* __restrict__ Wih1,
                const float* __restrict__ b1,  const float* __restrict__ gih1,
                const float* __restrict__ bih1, const float* __restrict__ ghh1,
                const float* __restrict__ bhh1, const float* __restrict__ gc1,
                const float* __restrict__ bc1,
                const float* __restrict__ b2,  const float* __restrict__ gih2,
                const float* __restrict__ bih2, const float* __restrict__ ghh2,
                const float* __restrict__ bhh2, const float* __restrict__ gc2,
                const float* __restrict__ bc2,
                char* __restrict__ ws, float* __restrict__ out)
{
    __shared__ float s_hh[G4];
    __shared__ float s_ih[G4];
    __shared__ float s_red[NTHR];

    unsigned* cnt = (unsigned*)(ws + OFF_CNT);
    const unsigned short* Whh1bf = (const unsigned short*)(ws + OFF_WHH1);
    const unsigned short* Wih2bf = (const unsigned short*)(ws + OFF_WIH2);
    const unsigned short* Whh2bf = (const unsigned short*)(ws + OFF_WHH2);
    unsigned short* h1bf = (unsigned short*)(ws + OFF_H1);
    unsigned short* h2bf = (unsigned short*)(ws + OFF_H2);
    float* c1   = (float*)(ws + OFF_C1);
    float* c2   = (float*)(ws + OFF_C2);
    float* raw1 = (float*)(ws + OFF_RAW1);
    float* r2ih = (float*)(ws + OFF_R2IH);
    float* r2hh = (float*)(ws + OFF_R2HH);

    const int lane = threadIdx.x & 31;
    const int wid  = threadIdx.x >> 5;

    // full-grid mapping: one 16x16 tile per wave (2048 waves = 8x256 tiles)
    const int wave = blockIdx.x * (NTHR / 32) + wid;
    const int m0 = (wave >> 8) * 16;
    const int n0 = (wave & 255) * 16;

    // half-grid mapping (WGs 128..255, 1024 waves): 16x32 tile pair per wave,
    // same row band so the A fragment is shared between the two WMMAs
    const int wl  = (blockIdx.x - Bdim) * (NTHR / 32) + wid;
    const int m0h = (wl >> 7) * 16;
    const int n0h = (wl & 127) * 32;

    unsigned phase = 0;

    // prologue: raw1 for t=0
    wmma_tile_bf16(h1bf, Whh1bf, raw1, m0, n0, lane);
    grid_sync(cnt, ++phase);

    for (int t = 0; t < Tdim; ++t) {
        // ---- P2: layer-1 cell || hidden GEMM r2hh(h2_{t-1}) ----
        if (blockIdx.x < Bdim) {
            const int b = blockIdx.x;
            lstm_elemwise(b, raw1, nullptr, x[(size_t)b * Tdim + t], Wih1,
                          gih1, bih1, ghh1, bhh1, b1, gc1, bc1,
                          c1, h1bf, nullptr, s_hh, s_ih, s_red);
        } else {
            wmma_tile_bf16_x2(h2bf, Whh2bf, r2hh, m0h, n0h, lane);
        }
        grid_sync(cnt, ++phase);

        // ---- P3: r2ih = h1_t @ Wih2^T (all WGs, on the critical path) ----
        wmma_tile_bf16(h1bf, Wih2bf, r2ih, m0, n0, lane);
        grid_sync(cnt, ++phase);

        // ---- P4: layer-2 cell + out accum || hidden GEMM raw1(t+1) = h1_t @ Whh1^T ----
        if (blockIdx.x < Bdim) {
            const int b = blockIdx.x;
            lstm_elemwise(b, r2hh, r2ih, 0.0f, nullptr,
                          gih2, bih2, ghh2, bhh2, b2, gc2, bc2,
                          c2, h2bf, out, s_hh, s_ih, s_red);
        } else {
            wmma_tile_bf16_x2(h1bf, Whh1bf, raw1, m0h, n0h, lane);
        }
        grid_sync(cnt, ++phase);
    }
}

// ---------------- entry point ----------------
extern "C" void kernel_launch(void* const* d_in, const int* in_sizes, int n_in,
                              void* d_out, int out_size, void* d_ws, size_t ws_size,
                              hipStream_t stream)
{
    const float* x    = (const float*)d_in[0];
    const float* Wih1 = (const float*)d_in[1];
    const float* Whh1 = (const float*)d_in[2];
    const float* b1   = (const float*)d_in[3];
    const float* gih1 = (const float*)d_in[4];
    const float* bih1 = (const float*)d_in[5];
    const float* ghh1 = (const float*)d_in[6];
    const float* bhh1 = (const float*)d_in[7];
    const float* gc1  = (const float*)d_in[8];
    const float* bc1  = (const float*)d_in[9];
    const float* Wih2 = (const float*)d_in[10];
    const float* Whh2 = (const float*)d_in[11];
    const float* b2   = (const float*)d_in[12];
    const float* gih2 = (const float*)d_in[13];
    const float* bih2 = (const float*)d_in[14];
    const float* ghh2 = (const float*)d_in[15];
    const float* bhh2 = (const float*)d_in[16];
    const float* gc2  = (const float*)d_in[17];
    const float* bc2  = (const float*)d_in[18];
    const float* h01  = (const float*)d_in[19];
    const float* c01  = (const float*)d_in[20];
    const float* h02  = (const float*)d_in[21];
    const float* c02  = (const float*)d_in[22];

    float* out = (float*)d_out;
    char*  ws  = (char*)d_ws;

    hipLaunchKernelGGL(init_kernel, dim3(2048), dim3(256), 0, stream,
                       Whh1, Wih2, Whh2, h01, c01, h02, c02, ws, out);

    hipLaunchKernelGGL(lstm_persistent, dim3(NBLK), dim3(NTHR), 0, stream,
                       x, Wih1, b1, gih1, bih1, ghh1, bhh1, gc1, bc1,
                       b2, gih2, bih2, ghh2, bhh2, gc2, bc2, ws, out);
}
